// Decoder_67095979099038
// MI455X (gfx1250) — matrix operand order repaired and checked by reading, not compiled, not run on hardware
//
#include <hip/hip_runtime.h>
#include <math.h>

#define Hd   512
#define BATCH 256
#define SEQ  512
#define G4   2048   // 4*H

typedef __attribute__((ext_vector_type(16))) __bf16 v16bf;
typedef __attribute__((ext_vector_type(8)))  float  v8f;

union Frag16 { unsigned int u[8]; v16bf v; };

static __device__ __forceinline__ unsigned short f2bf(float f) {
    unsigned int x = __float_as_uint(f);
    unsigned int r = (x + 0x7fffu + ((x >> 16) & 1u)) >> 16;   // RNE
    return (unsigned short)r;
}

static __device__ __forceinline__ float dtanh(float x) {
#if __has_builtin(__builtin_amdgcn_tanhf)
    return __builtin_amdgcn_tanhf(x);        // gfx1250 v_tanh_f32
#else
    float e = __expf(-2.f * fabsf(x));
    float t = (1.f - e) / (1.f + e);
    return copysignf(t, x);
#endif
}
static __device__ __forceinline__ float dsigmoid(float x) {
    return 1.f / (1.f + __expf(-x));
}

// ---------------------------------------------------------------------------
// K0: hb[b][g] = sum_k h[b][k] * Wa[g][k]  + ba[g]      (Wa_h = Wa[:, :H])
// ---------------------------------------------------------------------------
__global__ __launch_bounds__(256) void k_hb(const float* __restrict__ hidden,
                                            const float* __restrict__ Wa,
                                            const float* __restrict__ ba,
                                            float* __restrict__ hb) {
    __shared__ float sH[Hd];
    const int b = blockIdx.x, t = threadIdx.x;
    sH[t]       = hidden[(size_t)b * Hd + t];
    sH[t + 256] = hidden[(size_t)b * Hd + t + 256];
    __syncthreads();
    #pragma unroll
    for (int gi = 0; gi < 2; ++gi) {
        const int g = t + gi * 256;
        float acc = ba[g];
        const float* wr = Wa + (size_t)g * (2 * Hd);
        for (int k = 0; k < Hd; k += 4) {
            float4 w = *(const float4*)(wr + k);
            acc += sH[k] * w.x + sH[k+1] * w.y + sH[k+2] * w.z + sH[k+3] * w.w;
        }
        hb[(size_t)b * Hd + g] = acc;
    }
}

// ---------------------------------------------------------------------------
// KC: repack W_ih / W_hh to bf16 for the gates WMMA (split W_ih column 0)
// ---------------------------------------------------------------------------
__global__ __launch_bounds__(256) void k_convert(const float* __restrict__ W_ih,
                                                 const float* __restrict__ W_hh,
                                                 unsigned short* __restrict__ wih,
                                                 unsigned short* __restrict__ whh,
                                                 float* __restrict__ wi0) {
    const int tid = blockIdx.x * 256 + threadIdx.x;       // 2048*512 threads
    const int n = tid >> 9, k = tid & 511;
    wih[tid] = f2bf(W_ih[(size_t)n * 513 + 1 + k]);
    whh[tid] = f2bf(W_hh[tid]);
    if (k == 0) wi0[n] = W_ih[(size_t)n * 513];
}

// ---------------------------------------------------------------------------
// K1: fused  scores[b,s] = sum_g tanh( (E @ Wa_e^T)[bs,g] + hb[b,g] ) * v[g]
//     bf16 WMMA, energy tensor never materialized.
//     Workgroup = 8 waves x 16 rows = 128 rows of (b*S+s). Grid = 1024.
//     All 16 A k-fragments (16x512 bf16 per wave) are register-resident:
//     steady-state inner loop is 2x ds_load_b128 + 1x v_wmma.
// ---------------------------------------------------------------------------
__global__ __launch_bounds__(256) void k_attn_scores(const float* __restrict__ enc,
                                                     const float* __restrict__ Wa,
                                                     const float* __restrict__ hb,
                                                     const float* __restrict__ v,
                                                     float* __restrict__ scores) {
    __shared__ __align__(16) unsigned short sA[128 * 514];  // E tile, pad 2 (+1 bank/row)
    __shared__ __align__(16) unsigned short sB[16 * 520];   // Wa_e tile, pad 8 (+4 banks/row)

    const int t = threadIdx.x;
    const int lane = t & 31, wv = t >> 5;
    const int hi = lane >> 4, ln = lane & 15;
    const int m0 = blockIdx.x * 128;     // first global row (b*S + s)
    const int b  = m0 >> 9;              // 128-row block never crosses a batch

    // stage A: 128 x 512 fp32 -> bf16 LDS (row-major, padded stride 514)
    #pragma unroll 4
    for (int i = 0; i < 64; ++i) {
        const int q = i * 256 + t;
        const int row = q >> 7, col = (q & 127) << 2;
        const float4 f = *(const float4*)(enc + (size_t)(m0 + row) * Hd + col);
        *(unsigned int*)(sA + row * 514 + col)     = (unsigned int)f2bf(f.x) | ((unsigned int)f2bf(f.y) << 16);
        *(unsigned int*)(sA + row * 514 + col + 2) = (unsigned int)f2bf(f.z) | ((unsigned int)f2bf(f.w) << 16);
    }
    __syncthreads();

    // hoist ALL A fragments into registers: 16 kt x 8 VGPRs = 128 VGPRs/lane.
    // A 16x32 bf16 fragment, contiguous k order (as B): pair j -> K = 16*hi + 2j
    v16bf fa[16];
    {
        const unsigned short* ra = sA + (wv * 16 + ln) * 514;
        #pragma unroll
        for (int kt = 0; kt < 16; ++kt) {
            Frag16 f;
            #pragma unroll
            for (int j = 0; j < 8; ++j) {
                const int kk = kt * 32 + 16 * hi + 2 * j;
                f.u[j] = *(const unsigned int*)(ra + kk);
            }
            fa[kt] = f.v;
        }
    }

    float sc[8];
    #pragma unroll
    for (int r = 0; r < 8; ++r) sc[r] = 0.f;

    const unsigned short* rb = sB + ln * 520;

    for (int nt = 0; nt < 32; ++nt) {
        __syncthreads();
        // stage B tile: Wa_e rows g = nt*16 .. +15  (Wa_e[g][k] = Wa[g*1024 + 512 + k])
        #pragma unroll
        for (int i = 0; i < 8; ++i) {
            const int q = i * 256 + t;
            const int row = q >> 7, col = (q & 127) << 2;
            const float4 f = *(const float4*)(Wa + (size_t)(nt * 16 + row) * (2 * Hd) + Hd + col);
            *(unsigned int*)(sB + row * 520 + col)     = (unsigned int)f2bf(f.x) | ((unsigned int)f2bf(f.y) << 16);
            *(unsigned int*)(sB + row * 520 + col + 2) = (unsigned int)f2bf(f.z) | ((unsigned int)f2bf(f.w) << 16);
        }
        // speculative prefetch of next nt's Wa_e tile (one 128B line per thread)
        if (nt < 31) {
            const int prow = t >> 4, pcol = (t & 15) << 5;
            __builtin_prefetch(Wa + (size_t)((nt + 1) * 16 + prow) * (2 * Hd) + Hd + pcol, 0, 3);
        }
        __syncthreads();

        v8f c;
        #pragma unroll
        for (int r = 0; r < 8; ++r) c[r] = 0.f;

        #pragma unroll
        for (int kt = 0; kt < 16; ++kt) {
            // B 32x16 bf16: lane ln = column, halves hold K 0..15 / 16..31 contiguous
            Frag16 fb;
            const uint4* pb = (const uint4*)(rb + kt * 32 + hi * 16);
            const uint4 q0 = pb[0], q1 = pb[1];
            fb.u[0] = q0.x; fb.u[1] = q0.y; fb.u[2] = q0.z; fb.u[3] = q0.w;
            fb.u[4] = q1.x; fb.u[5] = q1.y; fb.u[6] = q1.z; fb.u[7] = q1.w;
            c = __builtin_amdgcn_wmma_f32_16x16x32_bf16(false, fa[kt], false, fb.v,
                                                        (short)0, c, false, false);
        }

        // fused epilogue: tanh(energy)*v accumulated per row
        const int n = nt * 16 + ln;
        const float bias = hb[(size_t)b * Hd + n];
        const float vn = v[n];
        #pragma unroll
        for (int r = 0; r < 8; ++r) sc[r] += dtanh(c[r] + bias) * vn;
    }

    // reduce over the 16 lanes that share each row
    #pragma unroll
    for (int r = 0; r < 8; ++r) {
        float x = sc[r];
        x += __shfl_xor(x, 8, 32);
        x += __shfl_xor(x, 4, 32);
        x += __shfl_xor(x, 2, 32);
        x += __shfl_xor(x, 1, 32);
        sc[r] = x;
    }
    if (ln == 0) {
        const int base = m0 + wv * 16 + hi * 8;   // C rows: M = hi*8 + r
        #pragma unroll
        for (int r = 0; r < 8; ++r) scores[base + r] = sc[r];
    }
}

// ---------------------------------------------------------------------------
// K2: softmax over S per batch row
// ---------------------------------------------------------------------------
__global__ __launch_bounds__(256) void k_softmax(const float* __restrict__ scores,
                                                 float* __restrict__ attn) {
    __shared__ float red[256];
    const int b = blockIdx.x, t = threadIdx.x;
    const float s0 = scores[b * SEQ + t], s1 = scores[b * SEQ + t + 256];
    red[t] = fmaxf(s0, s1);
    __syncthreads();
    for (int off = 128; off > 0; off >>= 1) {
        if (t < off) red[t] = fmaxf(red[t], red[t + off]);
        __syncthreads();
    }
    const float m = red[0];
    __syncthreads();
    const float e0 = __expf(s0 - m), e1 = __expf(s1 - m);
    red[t] = e0 + e1;
    __syncthreads();
    for (int off = 128; off > 0; off >>= 1) {
        if (t < off) red[t] += red[t + off];
        __syncthreads();
    }
    const float inv = 1.f / red[0];
    attn[b * SEQ + t]       = e0 * inv;
    attn[b * SEQ + t + 256] = e1 * inv;
}

// ---------------------------------------------------------------------------
// K3: context[b][h] = sum_s attn[b][s] * E[b][s][h]   (coalesced E stream)
// ---------------------------------------------------------------------------
__global__ __launch_bounds__(256) void k_context(const float* __restrict__ attn,
                                                 const float* __restrict__ enc,
                                                 float* __restrict__ ctx) {
    __shared__ float sw[SEQ];
    const int b = blockIdx.x, t = threadIdx.x;
    const int h = blockIdx.y * 256 + t;
    sw[t]       = attn[b * SEQ + t];
    sw[t + 256] = attn[b * SEQ + t + 256];
    __syncthreads();
    float acc = 0.f;
    const float* er = enc + (size_t)b * SEQ * Hd + h;
    #pragma unroll 8
    for (int s = 0; s < 512; ++s) {
        if ((s & 63) == 0 && s + 64 < 512)
            __builtin_prefetch(er + (size_t)(s + 64) * Hd, 0, 0);  // speculative stream-ahead
        acc += sw[s] * er[(size_t)s * Hd];
    }
    ctx[(size_t)b * Hd + h] = acc;
}

// ---------------------------------------------------------------------------
// K4: gates = [ctx||h](256x1024) @ [Wih'|Whh](2048x1024)^T + biases + input*Wih[:,0]
//     bf16 WMMA. Grid (16 M-tiles, 16 N-groups), 8 waves x 16 cols each.
// ---------------------------------------------------------------------------
__global__ __launch_bounds__(256) void k_gates(const float* __restrict__ ctx,
                                               const float* __restrict__ hidden,
                                               const unsigned short* __restrict__ wih,
                                               const unsigned short* __restrict__ whh,
                                               const float* __restrict__ wi0,
                                               const float* __restrict__ input,
                                               const float* __restrict__ b_ih,
                                               const float* __restrict__ b_hh,
                                               float* __restrict__ gates) {
    __shared__ __align__(16) unsigned short sX[16 * 1028];   // [ctx||h] tile, padded
    const int t = threadIdx.x, lane = t & 31, wv = t >> 5;
    const int hi = lane >> 4, ln = lane & 15;
    const int m0 = blockIdx.x * 16;
    const int n0 = (blockIdx.y * 8 + wv) * 16;

    #pragma unroll 4
    for (int i = 0; i < 64; ++i) {
        const int e = i * 256 + t;
        const int row = e >> 10, col = e & 1023;
        const float val = (col < Hd) ? ctx[(size_t)(m0 + row) * Hd + col]
                                     : hidden[(size_t)(m0 + row) * Hd + (col - Hd)];
        sX[row * 1028 + col] = f2bf(val);
    }
    __syncthreads();

    v8f c;
    #pragma unroll
    for (int r = 0; r < 8; ++r) c[r] = 0.f;

    const unsigned short* ra = sX + ln * 1028;
    const int n = n0 + ln;

    #pragma unroll 2
    for (int kt = 0; kt < 32; ++kt) {
        Frag16 fa, fb;
        #pragma unroll
        for (int j = 0; j < 8; ++j) {
            const int kk = kt * 32 + 16 * hi + 2 * j;
            fa.u[j] = *(const unsigned int*)(ra + kk);
        }
        const unsigned short* wb = (kt < 16)
            ? (wih + (size_t)n * 512 + kt * 32 + hi * 16)
            : (whh + (size_t)n * 512 + (kt - 16) * 32 + hi * 16);
        const uint4 q0 = *(const uint4*)(wb);
        const uint4 q1 = *(const uint4*)(wb + 8);
        fb.u[0] = q0.x; fb.u[1] = q0.y; fb.u[2] = q0.z; fb.u[3] = q0.w;
        fb.u[4] = q1.x; fb.u[5] = q1.y; fb.u[6] = q1.z; fb.u[7] = q1.w;
        c = __builtin_amdgcn_wmma_f32_16x16x32_bf16(false, fa.v, false, fb.v,
                                                    (short)0, c, false, false);
    }

    const float bi = b_ih[n] + b_hh[n];
    const float w0 = wi0[n];
    #pragma unroll
    for (int r = 0; r < 8; ++r) {
        const int brow = m0 + hi * 8 + r;
        gates[(size_t)brow * G4 + n] = c[r] + bi + input[brow] * w0;
    }
}

// ---------------------------------------------------------------------------
// K5: LSTM pointwise cell + h_new + c_new + prediction (fc reduce)
// ---------------------------------------------------------------------------
__global__ __launch_bounds__(256) void k_cell(const float* __restrict__ gates,
                                              const float* __restrict__ cell,
                                              const float* __restrict__ fc_W,
                                              const float* __restrict__ fc_b,
                                              float* __restrict__ pred,
                                              float* __restrict__ h_new,
                                              float* __restrict__ c_new) {
    __shared__ float red[256];
    const int b = blockIdx.x, t = threadIdx.x;
    const float* g = gates + (size_t)b * G4;
    float acc = 0.f;
    #pragma unroll
    for (int gi = 0; gi < 2; ++gi) {
        const int j = t + gi * 256;
        const float ii = dsigmoid(g[j]);
        const float ff = dsigmoid(g[j + 512]);
        const float gg = dtanh(g[j + 1024]);
        const float oo = dsigmoid(g[j + 1536]);
        const float cn = ff * cell[(size_t)b * Hd + j] + ii * gg;
        const float hn = oo * dtanh(cn);
        c_new[(size_t)b * Hd + j] = cn;
        h_new[(size_t)b * Hd + j] = hn;
        acc += hn * fc_W[j];
    }
    red[t] = acc;
    __syncthreads();
    for (int off = 128; off > 0; off >>= 1) {
        if (t < off) red[t] += red[t + off];
        __syncthreads();
    }
    if (t == 0) pred[b] = red[0] + fc_b[0];
}

// ---------------------------------------------------------------------------
extern "C" void kernel_launch(void* const* d_in, const int* in_sizes, int n_in,
                              void* d_out, int out_size, void* d_ws, size_t ws_size,
                              hipStream_t stream) {
    const float* input  = (const float*)d_in[0];
    const float* hidden = (const float*)d_in[1];
    const float* cell   = (const float*)d_in[2];
    const float* enc    = (const float*)d_in[3];
    const float* Wa     = (const float*)d_in[4];
    const float* ba     = (const float*)d_in[5];
    const float* v      = (const float*)d_in[6];
    const float* W_ih   = (const float*)d_in[7];
    const float* W_hh   = (const float*)d_in[8];
    const float* b_ih   = (const float*)d_in[9];
    const float* b_hh   = (const float*)d_in[10];
    const float* fc_W   = (const float*)d_in[11];
    const float* fc_b   = (const float*)d_in[12];

    float* out   = (float*)d_out;
    float* pred  = out;                         // (B,1)
    float* h_new = out + 256;                   // (1,B,H)
    float* c_new = out + 256 + 131072;          // (1,B,H)
    float* attn  = out + 256 + 2 * 131072;      // (B,S)

    char* ws = (char*)d_ws;
    float*          hb     = (float*)(ws + 0);          // B*H
    float*          scores = (float*)(ws + 524288);     // B*S
    float*          ctx    = (float*)(ws + 1048576);    // B*H
    float*          gatesb = (float*)(ws + 1572864);    // B*4H
    unsigned short* wih    = (unsigned short*)(ws + 3670016);  // 2048x512 bf16
    unsigned short* whh    = (unsigned short*)(ws + 5767168);  // 2048x512 bf16
    float*          wi0    = (float*)(ws + 7864320);    // 2048

    k_hb<<<256, 256, 0, stream>>>(hidden, Wa, ba, hb);
    k_convert<<<4096, 256, 0, stream>>>(W_ih, W_hh, wih, whh, wi0);
    k_attn_scores<<<1024, 256, 0, stream>>>(enc, Wa, hb, v, scores);
    k_softmax<<<256, 256, 0, stream>>>(scores, attn);
    k_context<<<dim3(256, 2), 256, 0, stream>>>(attn, enc, ctx);
    k_gates<<<dim3(16, 16), 256, 0, stream>>>(ctx, hidden, wih, whh, wi0,
                                              input, b_ih, b_hh, gatesb);
    k_cell<<<256, 256, 0, stream>>>(gatesb, cell, fc_W, fc_b, pred, h_new, c_new);
}
